// GCPNet_2946347565441
// MI455X (gfx1250) — matrix operand order repaired and checked by reading, not compiled, not run on hardware
//
#include <hip/hip_runtime.h>
#include <stdint.h>

// ---------------------------------------------------------------------------
// GCPNet forward for MI455X (gfx1250): bf16 WMMA GEMMs + fp32 VALU LSTM/LN.
// N=4096 nodes, D=64, TMAX=32 steps, 8 graphs of 512 nodes.
// ---------------------------------------------------------------------------

#define NN    4096
#define DD    64
#define TMAX  32
#define SLICE 512

typedef __attribute__((ext_vector_type(16))) __bf16 v16bf;
typedef __attribute__((ext_vector_type(8)))  float  v8f;

// fp32 -> bf16 bits, round-to-nearest-even (bit-exact, no cast codegen reliance)
__device__ __forceinline__ unsigned short f2bf_bits(float f) {
    union { float f; unsigned u; } x; x.f = f;
    unsigned r = x.u + 0x7fffu + ((x.u >> 16) & 1u);
    return (unsigned short)(r >> 16);
}
__device__ __forceinline__ __bf16 f2bf(float f) {
    unsigned short h = f2bf_bits(f);
    __bf16 b; __builtin_memcpy(&b, &h, 2); return b;
}
__device__ __forceinline__ unsigned pack2bf(float lo, float hi) {
    return (unsigned)f2bf_bits(lo) | ((unsigned)f2bf_bits(hi) << 16);
}

__device__ __forceinline__ float sigm(float x) { return 1.0f / (1.0f + __expf(-x)); }

__device__ __forceinline__ float wave_sum32(float v) {
    #pragma unroll
    for (int m = 16; m > 0; m >>= 1) v += __shfl_xor(v, m, 32);
    return v;
}

// ---------------------------------------------------------------------------
// bf16 WMMA GEMM:  Out[M x Ncols](f32) = A[M x Kdim] * Bt[Ncols x Kdim]^T
// A row-major bf16 (K contiguous). Bt is [Ncols][Kdim] bf16 (K contiguous).
// One wave per block computes a 32x64 tile (2 M-tiles x 4 N-tiles = 8 acc):
// each B fragment feeds 2 WMMAs and all 12 fragment loads of a k-chunk are
// issued before the 8 WMMAs -> staged s_wait_loadcnt instead of full drains.
// Kdim % 32 == 0.  grid = (M/32, Ncols/64), block = 32.
// ---------------------------------------------------------------------------
__global__ void __launch_bounds__(32) k_wmma_gemm(
    const __bf16* __restrict__ A, const __bf16* __restrict__ Bt,
    float* __restrict__ Out, int Kdim, int Ncols)
{
    const int lane = threadIdx.x & 31;
    const int hf   = lane >> 4;      // which 16-lane half
    const int l16  = lane & 15;
    const int row0 = blockIdx.x * 32;
    const int col0 = blockIdx.y * 64;

    const v8f vzero = {0.f,0.f,0.f,0.f,0.f,0.f,0.f,0.f};
    v8f acc[2][4];
    #pragma unroll
    for (int mt = 0; mt < 2; ++mt)
        #pragma unroll
        for (int nt = 0; nt < 4; ++nt) acc[mt][nt] = vzero;

    const __bf16* arow0 = A + (size_t)(row0 + l16) * Kdim;
    const __bf16* arow1 = A + (size_t)(row0 + 16 + l16) * Kdim;

    for (int k0 = 0; k0 < Kdim; k0 += 32) {
        // ---- load phase: 4x b128 (A) + 8x b128 (B), then compute phase ----
        // A fragment: elems 0..7  -> K = k0 + 8*hf + 0..7
        //             elems 8..15 -> K = k0 + 16 + 8*hf + 0..7
        v16bf a0, a1;
        {
            const __bf16* ap0 = arow0 + k0 + 8 * hf;
            const __bf16* ap1 = arow1 + k0 + 8 * hf;
            ((uint4*)&a0)[0] = *(const uint4*)(const void*)(ap0);
            ((uint4*)&a0)[1] = *(const uint4*)(const void*)(ap0 + 16);
            ((uint4*)&a1)[0] = *(const uint4*)(const void*)(ap1);
            ((uint4*)&a1)[1] = *(const uint4*)(const void*)(ap1 + 16);
        }
        // B fragment: same K order as A (elems 0..7 -> k0+8*hf+e, 8..15 -> k0+16+8*hf+e-8)
        v16bf bf[4];
        #pragma unroll
        for (int nt = 0; nt < 4; ++nt) {
            const __bf16* bp = Bt + (size_t)(col0 + nt * 16 + l16) * Kdim + k0 + 8 * hf;
            ((uint4*)&bf[nt])[0] = *(const uint4*)(const void*)(bp);
            ((uint4*)&bf[nt])[1] = *(const uint4*)(const void*)(bp + 16);
        }
        if (k0 + 32 < Kdim) {                    // gfx1250 global_prefetch_b8
            __builtin_prefetch(arow0 + k0 + 32, 0, 1);
            __builtin_prefetch(arow1 + k0 + 32, 0, 1);
        }
        #pragma unroll
        for (int nt = 0; nt < 4; ++nt) {
            acc[0][nt] = __builtin_amdgcn_wmma_f32_16x16x32_bf16(
                false, a0, false, bf[nt], (short)0, acc[0][nt], false, false);
            acc[1][nt] = __builtin_amdgcn_wmma_f32_16x16x32_bf16(
                false, a1, false, bf[nt], (short)0, acc[1][nt], false, false);
        }
    }

    // C/D layout: VGPR r -> M = row0 + mt*16 + r + 8*hf, N = col0 + nt*16 + l16
    #pragma unroll
    for (int mt = 0; mt < 2; ++mt)
        #pragma unroll
        for (int nt = 0; nt < 4; ++nt)
            #pragma unroll
            for (int r = 0; r < 8; ++r)
                Out[(size_t)(row0 + mt * 16 + r + 8 * hf) * Ncols
                    + col0 + nt * 16 + l16] = acc[mt][nt][r];
}

// ---------------------------------------------------------------------------
// Setup / conversion kernels (bf16 stores packed 2-per-dword)
// ---------------------------------------------------------------------------
__global__ void k_f32_to_bf16x2(const float* __restrict__ src,
                                unsigned* __restrict__ dst, int n2) {
    for (int i = blockIdx.x * blockDim.x + threadIdx.x; i < n2; i += gridDim.x * blockDim.x)
        dst[i] = pack2bf(src[2 * i], src[2 * i + 1]);
}

// Wcomb[256][192] = [lstm_v.Wx (256x128) | lstm_v.Wh (256x64)], bf16
__global__ void k_build_wcomb(const float* __restrict__ Wx, const float* __restrict__ Wh,
                              __bf16* __restrict__ W) {
    int i = blockIdx.x * blockDim.x + threadIdx.x;
    if (i >= 256 * 192) return;
    int g = i / 192, j = i % 192;
    float v = (j < 128) ? Wx[g * 128 + j] : Wh[g * 64 + (j - 128)];
    W[i] = f2bf(v);
}

__global__ void k_init(const float* __restrict__ v_normal, const float* __restrict__ c0,
                       float* __restrict__ Vh, float* __restrict__ Vc,
                       float* __restrict__ Ch, float* __restrict__ Cc, float invscale) {
    int i = blockIdx.x * blockDim.x + threadIdx.x;
    if (i < NN * DD) { Vh[i] = v_normal[i & 63] * invscale; Vc[i] = 0.0f; }
    if (i < DD)      { Ch[i] = c0[i]; Cc[i] = 0.0f; }
}

// VhT[d][i] = bf16(Vh[i][d])  (K-contiguous B operand). Packed: 2 nodes/thread.
__global__ void k_transpose_vh(const float* __restrict__ Vh, unsigned* __restrict__ VhT2) {
    int idx2 = blockIdx.x * blockDim.x + threadIdx.x;     // over DD*NN/2
    if (idx2 >= DD * NN / 2) return;
    int idx = idx2 * 2;
    int d = idx / NN, i = idx % NN;                        // NN even -> same d for i,i+1
    VhT2[idx2] = pack2bf(Vh[i * DD + d], Vh[(i + 1) * DD + d]);
}

// X[i][0:64]=Mv[i], X[i][64:128]=KC, X[i][128:192]=Vh[i]. Packed: 2 elems/thread.
__global__ void k_build_x(const float* __restrict__ Mv, const float* __restrict__ KC,
                          const float* __restrict__ Vh, unsigned* __restrict__ X2) {
    int idx2 = blockIdx.x * blockDim.x + threadIdx.x;     // over NN*192/2
    if (idx2 >= NN * 96) return;
    int idx = idx2 * 2;
    int i = idx / 192, j = idx % 192;                      // 192 even -> same row, j even
    float v0, v1;
    if (j < 64)       { v0 = Mv[i * 64 + j];        v1 = Mv[i * 64 + j + 1]; }
    else if (j < 128) { v0 = KC[j - 64];            v1 = KC[j - 63]; }
    else              { v0 = Vh[i * 64 + (j - 128)]; v1 = Vh[i * 64 + (j - 127)]; }
    X2[idx2] = pack2bf(v0, v1);
}

// ---------------------------------------------------------------------------
// KC = K * mlpC(Ch)   (single block, 64 threads)
// ---------------------------------------------------------------------------
__global__ void __launch_bounds__(64) k_mlpC(
    const float* __restrict__ Ch,
    const float* __restrict__ W1, const float* __restrict__ b1,
    const float* __restrict__ W2, const float* __restrict__ b2,
    float Kmul, float* __restrict__ KC)
{
    __shared__ float sh[64], sh1[64];
    int t = threadIdx.x;
    sh[t] = Ch[t];
    __syncthreads();
    float a = b1[t];
    for (int j = 0; j < 64; ++j) a += W1[t * 64 + j] * sh[j];
    sh1[t] = fmaxf(a, 0.0f);
    __syncthreads();
    float c = b2[t];
    for (int j = 0; j < 64; ++j) c += W2[t * 64 + j] * sh1[j];
    KC[t] = Kmul * c;
}

// ---------------------------------------------------------------------------
// LayerNorm-LSTM (variable side) + mlpV partial reduction.
// One wave per node (lane owns dims d and d+32); block = 8 nodes.
// Z holds the 4096x256 gate pre-activations (bias not yet added).
// ---------------------------------------------------------------------------
__global__ void __launch_bounds__(256) k_lstm_v(
    const float* __restrict__ Z,    const float* __restrict__ bias,
    const float* __restrict__ ln_g, const float* __restrict__ ln_b,
    const float* __restrict__ lnc_g,const float* __restrict__ lnc_b,
    const float* __restrict__ W1,   const float* __restrict__ b1,
    const float* __restrict__ W2,   const float* __restrict__ b2,
    float* __restrict__ Vh, float* __restrict__ Vc, float* __restrict__ partials)
{
    __shared__ float shA[8][64];
    __shared__ float shB[8][64];
    const int warp = threadIdx.x >> 5;
    const int lane = threadIdx.x & 31;
    const int node = blockIdx.x * 8 + warp;
    const int d0 = lane, d1 = lane + 32;

    float act0[4], act1[4];
    #pragma unroll
    for (int g = 0; g < 4; ++g) {
        float z0 = Z[(size_t)node * 256 + g * 64 + d0] + bias[g * 64 + d0];
        float z1 = Z[(size_t)node * 256 + g * 64 + d1] + bias[g * 64 + d1];
        float mean = wave_sum32(z0 + z1) * (1.0f / 64.0f);
        float e0 = z0 - mean, e1 = z1 - mean;
        float var = wave_sum32(e0 * e0 + e1 * e1) * (1.0f / 64.0f);
        float inv = rsqrtf(var + 1e-5f);
        float y0 = e0 * inv * ln_g[g * 64 + d0] + ln_b[g * 64 + d0];
        float y1 = e1 * inv * ln_g[g * 64 + d1] + ln_b[g * 64 + d1];
        if (g == 2) { act0[g] = tanhf(y0); act1[g] = tanhf(y1); }
        else        { act0[g] = sigm(y0);  act1[g] = sigm(y1);  }
    }
    float c20 = act0[1] * Vc[(size_t)node * 64 + d0] + act0[0] * act0[2];
    float c21 = act1[1] * Vc[(size_t)node * 64 + d1] + act1[0] * act1[2];
    float mc  = wave_sum32(c20 + c21) * (1.0f / 64.0f);
    float e0 = c20 - mc, e1 = c21 - mc;
    float vv = wave_sum32(e0 * e0 + e1 * e1) * (1.0f / 64.0f);
    float inv = rsqrtf(vv + 1e-5f);
    float h20 = act0[3] * tanhf(e0 * inv * lnc_g[d0] + lnc_b[d0]);
    float h21 = act1[3] * tanhf(e1 * inv * lnc_g[d1] + lnc_b[d1]);

    Vh[(size_t)node * 64 + d0] = h20;  Vh[(size_t)node * 64 + d1] = h21;
    Vc[(size_t)node * 64 + d0] = c20;  Vc[(size_t)node * 64 + d1] = c21;

    // mlpV(Vh2): y = relu(h2 @ W1^T + b1) @ W2^T + b2  -> per-block partial sum
    shA[warp][d0] = h20; shA[warp][d1] = h21;
    __syncthreads();
    float h1a = b1[d0], h1b = b1[d1];
    for (int j = 0; j < 64; ++j) { float hv = shA[warp][j]; h1a += W1[d0 * 64 + j] * hv; h1b += W1[d1 * 64 + j] * hv; }
    h1a = fmaxf(h1a, 0.0f); h1b = fmaxf(h1b, 0.0f);
    shB[warp][d0] = h1a; shB[warp][d1] = h1b;
    __syncthreads();
    float ya = b2[d0], yb = b2[d1];
    for (int j = 0; j < 64; ++j) { float hv = shB[warp][j]; ya += W2[d0 * 64 + j] * hv; yb += W2[d1 * 64 + j] * hv; }
    __syncthreads();
    shA[warp][d0] = ya; shA[warp][d1] = yb;
    __syncthreads();
    if (threadIdx.x < 64) {
        float s = 0.0f;
        for (int w = 0; w < 8; ++w) s += shA[w][threadIdx.x];
        partials[blockIdx.x * 64 + threadIdx.x] = s;   // deterministic, no atomics
    }
}

// ---------------------------------------------------------------------------
// Clause update: xc = sum partials; LayerNorm-LSTM on the single clause state.
// Single block of 64 threads (thread t owns dim t).
// ---------------------------------------------------------------------------
__global__ void __launch_bounds__(64) k_clause(
    const float* __restrict__ partials, int nblocks,
    const float* __restrict__ Wx, const float* __restrict__ Wh, const float* __restrict__ bias,
    const float* __restrict__ ln_g, const float* __restrict__ ln_b,
    const float* __restrict__ lnc_g, const float* __restrict__ lnc_b,
    float* __restrict__ Ch, float* __restrict__ Cc)
{
    __shared__ float s_xc[64], s_ch[64], s_z[256], s_c2[64], s_m[5], s_v[5];
    int t = threadIdx.x;
    float s = 0.0f;
    for (int bb = 0; bb < nblocks; ++bb) s += partials[bb * 64 + t];
    s_xc[t] = s;
    s_ch[t] = Ch[t];
    __syncthreads();
    float zr[4];
    #pragma unroll
    for (int g = 0; g < 4; ++g) {
        int row = g * 64 + t;
        float a = bias[row];
        for (int j = 0; j < 64; ++j) a += Wx[row * 64 + j] * s_xc[j] + Wh[row * 64 + j] * s_ch[j];
        zr[g] = a; s_z[row] = a;
    }
    __syncthreads();
    if (t < 4) { float m = 0; for (int j = 0; j < 64; ++j) m += s_z[t * 64 + j]; s_m[t] = m * (1.0f / 64.0f); }
    __syncthreads();
    if (t < 4) { float v = 0; for (int j = 0; j < 64; ++j) { float d = s_z[t * 64 + j] - s_m[t]; v += d * d; } s_v[t] = v * (1.0f / 64.0f); }
    __syncthreads();
    float act[4];
    #pragma unroll
    for (int g = 0; g < 4; ++g) {
        float y = (zr[g] - s_m[g]) * rsqrtf(s_v[g] + 1e-5f) * ln_g[g * 64 + t] + ln_b[g * 64 + t];
        act[g] = (g == 2) ? tanhf(y) : sigm(y);
    }
    float c2 = act[1] * Cc[t] + act[0] * act[2];
    s_c2[t] = c2;
    __syncthreads();
    if (t == 0) { float m = 0; for (int j = 0; j < 64; ++j) m += s_c2[j]; s_m[4] = m * (1.0f / 64.0f); }
    __syncthreads();
    if (t == 0) { float v = 0; for (int j = 0; j < 64; ++j) { float d = s_c2[j] - s_m[4]; v += d * d; } s_v[4] = v * (1.0f / 64.0f); }
    __syncthreads();
    float h2 = act[3] * tanhf((c2 - s_m[4]) * rsqrtf(s_v[4] + 1e-5f) * lnc_g[t] + lnc_b[t]);
    Ch[t] = h2; Cc[t] = c2;
}

// ---------------------------------------------------------------------------
// Vote MLP: per-node scalar vote, per-block (8 nodes) partial sum.
// Blocks are 512/8 = 64 per graph, aligned to graph boundaries.
// ---------------------------------------------------------------------------
__global__ void __launch_bounds__(256) k_vote(
    const float* __restrict__ Vh,
    const float* __restrict__ W1, const float* __restrict__ b1,
    const float* __restrict__ W2, const float* __restrict__ b2,
    float* __restrict__ partialsV)
{
    __shared__ float shA[8][64];
    __shared__ float s_y[8];
    const int warp = threadIdx.x >> 5;
    const int lane = threadIdx.x & 31;
    const int node = blockIdx.x * 8 + warp;
    const int d0 = lane, d1 = lane + 32;

    shA[warp][d0] = Vh[(size_t)node * 64 + d0];
    shA[warp][d1] = Vh[(size_t)node * 64 + d1];
    __syncthreads();
    float h1a = b1[d0], h1b = b1[d1];
    for (int j = 0; j < 64; ++j) { float hv = shA[warp][j]; h1a += W1[d0 * 64 + j] * hv; h1b += W1[d1 * 64 + j] * hv; }
    h1a = fmaxf(h1a, 0.0f); h1b = fmaxf(h1b, 0.0f);
    float p = wave_sum32(W2[d0] * h1a + W2[d1] * h1b);
    if (lane == 0) s_y[warp] = p + b2[0];
    __syncthreads();
    if (threadIdx.x == 0) {
        float ss = 0.0f;
        for (int w = 0; w < 8; ++w) ss += s_y[w];
        partialsV[blockIdx.x] = ss;
    }
}

__global__ void k_copy(const float* __restrict__ src, float* __restrict__ dst, int n) {
    int i = blockIdx.x * blockDim.x + threadIdx.x;
    if (i < n) dst[i] = src[i];
}

// out[0:8]=pred, out[8:16]=means, out[16:16+NN*DD]=Vh (copied separately),
// out[16+NN*DD : +64]=Ch
__global__ void __launch_bounds__(64) k_final(
    const float* __restrict__ partialsV, const float* __restrict__ Ch, float* __restrict__ out)
{
    int t = threadIdx.x;
    if (t < 8) {
        float s = 0.0f;
        for (int bb = 0; bb < 64; ++bb) s += partialsV[t * 64 + bb];
        float mean = s * (1.0f / (float)SLICE);
        out[t]     = sigm(mean);
        out[8 + t] = mean;
    }
    out[16 + NN * DD + t] = Ch[t];
}

// ---------------------------------------------------------------------------
extern "C" void kernel_launch(void* const* d_in, const int* in_sizes, int n_in,
                              void* d_out, int out_size, void* d_ws, size_t ws_size,
                              hipStream_t stream) {
    // setup_inputs() flattening (depth-first dict order)
    const float* M_vv    = (const float*)d_in[0];
    const float* mlpV_W1 = (const float*)d_in[2];
    const float* mlpV_b1 = (const float*)d_in[3];
    const float* mlpV_W2 = (const float*)d_in[4];
    const float* mlpV_b2 = (const float*)d_in[5];
    const float* mlpC_W1 = (const float*)d_in[6];
    const float* mlpC_b1 = (const float*)d_in[7];
    const float* mlpC_W2 = (const float*)d_in[8];
    const float* mlpC_b2 = (const float*)d_in[9];
    const float* vote_W1 = (const float*)d_in[10];
    const float* vote_b1 = (const float*)d_in[11];
    const float* vote_W2 = (const float*)d_in[12];
    const float* vote_b2 = (const float*)d_in[13];
    const float* lv_Wx   = (const float*)d_in[14];
    const float* lv_Wh   = (const float*)d_in[15];
    const float* lv_b    = (const float*)d_in[16];
    const float* lv_lng  = (const float*)d_in[17];
    const float* lv_lnb  = (const float*)d_in[18];
    const float* lv_lncg = (const float*)d_in[19];
    const float* lv_lncb = (const float*)d_in[20];
    const float* lc_Wx   = (const float*)d_in[21];
    const float* lc_Wh   = (const float*)d_in[22];
    const float* lc_b    = (const float*)d_in[23];
    const float* lc_lng  = (const float*)d_in[24];
    const float* lc_lnb  = (const float*)d_in[25];
    const float* lc_lncg = (const float*)d_in[26];
    const float* lc_lncb = (const float*)d_in[27];
    const float* v_normal= (const float*)d_in[28];
    const float* c0      = (const float*)d_in[29];

    // Workspace carve (256B aligned slices)
    char* p = (char*)d_ws;
    auto carve = [&](size_t bytes) -> void* {
        void* r = (void*)p; p += (bytes + 255) & ~(size_t)255; return r;
    };
    __bf16* MvvBf   = (__bf16*)carve((size_t)NN * NN * 2);   // 32 MB, L2-resident
    __bf16* VhT     = (__bf16*)carve((size_t)DD * NN * 2);
    __bf16* Xbf     = (__bf16*)carve((size_t)NN * 192 * 2);
    __bf16* WcombBf = (__bf16*)carve((size_t)256 * 192 * 2);
    float*  Vh      = (float*)carve((size_t)NN * DD * 4);
    float*  Vc      = (float*)carve((size_t)NN * DD * 4);
    float*  Mv      = (float*)carve((size_t)NN * DD * 4);
    float*  Z       = (float*)carve((size_t)NN * 256 * 4);
    float*  partials= (float*)carve((size_t)512 * 64 * 4);
    float*  partialsV=(float*)carve((size_t)512 * 4);
    float*  KC      = (float*)carve(64 * 4);
    float*  Ch      = (float*)carve(64 * 4);
    float*  Cc      = (float*)carve(64 * 4);
    (void)ws_size; (void)n_in; (void)out_size;

    float Kmul     = (float)(in_sizes[1] / NN);              // M_vc.shape[1] = 1024
    float invscale = 1.0f / (float)((NN - 1) / 50 + 1);      // 1/82

    // One-time conversions (re-done every call: deterministic, no caching)
    k_f32_to_bf16x2<<<8192, 256, 0, stream>>>(M_vv, (unsigned*)MvvBf, NN * NN / 2);
    k_build_wcomb<<<(256 * 192 + 255) / 256, 256, 0, stream>>>(lv_Wx, lv_Wh, WcombBf);
    k_init<<<(NN * DD + 255) / 256, 256, 0, stream>>>(v_normal, c0, Vh, Vc, Ch, Cc, invscale);

    for (int step = 0; step < TMAX; ++step) {
        // C = mlpC(Ch); KC = K*C
        k_mlpC<<<1, 64, 0, stream>>>(Ch, mlpC_W1, mlpC_b1, mlpC_W2, mlpC_b2, Kmul, KC);
        // Mv = M_vv @ Vh  (bf16 WMMA; M_vv stays hot in 192MB L2)
        k_transpose_vh<<<(DD * NN / 2 + 255) / 256, 256, 0, stream>>>(Vh, (unsigned*)VhT);
        k_wmma_gemm<<<dim3(NN / 32, 1), 32, 0, stream>>>(MvvBf, VhT, Mv, NN, DD);
        // Z = [Mv | KC | Vh] @ [Wx | Wh]^T  (bf16 WMMA, K=192)
        k_build_x<<<(NN * 96 + 255) / 256, 256, 0, stream>>>(Mv, KC, Vh, (unsigned*)Xbf);
        k_wmma_gemm<<<dim3(NN / 32, 4), 32, 0, stream>>>(Xbf, WcombBf, Z, 192, 256);
        // LSTM_V + mlpV partials
        k_lstm_v<<<NN / 8, 256, 0, stream>>>(Z, lv_b, lv_lng, lv_lnb, lv_lncg, lv_lncb,
                                             mlpV_W1, mlpV_b1, mlpV_W2, mlpV_b2,
                                             Vh, Vc, partials);
        // xc reduce + LSTM_C
        k_clause<<<1, 64, 0, stream>>>(partials, NN / 8, lc_Wx, lc_Wh, lc_b,
                                       lc_lng, lc_lnb, lc_lncg, lc_lncb, Ch, Cc);
    }

    k_vote<<<NN / 8, 256, 0, stream>>>(Vh, vote_W1, vote_b1, vote_W2, vote_b2, partialsV);
    k_copy<<<(NN * DD + 255) / 256, 256, 0, stream>>>(Vh, (float*)d_out + 16, NN * DD);
    k_final<<<1, 64, 0, stream>>>(partialsV, Ch, (float*)d_out);
}
